// MeshGraphNetTransformer_14121852469813
// MI455X (gfx1250) — matrix operand order repaired and checked by reading, not compiled, not run on hardware
//
#include <hip/hip_runtime.h>
#include <hip/hip_bf16.h>
#include <cstdint>
#include <cstddef>

#define HID 128
#define NSLICES 32

typedef __attribute__((ext_vector_type(16))) _Float16 v16h;
typedef __attribute__((ext_vector_type(8)))  _Float16 v8h;
typedef __attribute__((ext_vector_type(8)))  float    v8f;

__device__ __forceinline__ float gelu_exact(float x) {
  return 0.5f * x * (1.0f + erff(x * 0.7071067811865476f));
}

// ---------------- CDNA5 async gather -> LDS (ASYNCcnt path) ------------------
// Per-lane: copies 16 B from global vaddr into LDS at vdst; low 32 bits of a generic
// pointer into __shared__ are the wave-relative LDS offset (flat-LDS aperture).
__device__ __forceinline__ void async_ld_b128(void* lds_dst, const void* gsrc) {
  uint32_t dst = (uint32_t)(uintptr_t)lds_dst;
  asm volatile("global_load_async_to_lds_b128 %0, %1, off"
               :
               : "v"(dst), "v"(gsrc)
               : "memory");
}
__device__ __forceinline__ void wait_async0() {
  asm volatile("s_wait_asynccnt 0x0" ::: "memory");
}

// ---------------- WMMA fragment helpers (wave32, v_wmma_f32_16x16x32_f16) ----
// A (16x32 f16): lane L -> row m0+(L&15); halves [0..7]=A[m][kb..kb+7],
// [8..15]=A[m][kb+16..kb+23], kb = k0 + 8*(L>>4).  (ISA 7.12.2)
__device__ __forceinline__ v16h load_a_frag(const _Float16* A, int lda, int m0,
                                            int k0, int lane, int Mclamp) {
  int m = m0 + (lane & 15);
  if (m >= Mclamp) m = Mclamp - 1;
  const _Float16* p = A + (size_t)m * lda + (k0 + 8 * (lane >> 4));
  v8h lo = *(const v8h*)(p);
  v8h hi = *(const v8h*)(p + 16);
  v16h r;
#pragma unroll
  for (int i = 0; i < 8; ++i) { r[i] = lo[i]; r[i + 8] = hi[i]; }
  return r;
}
// B (32x16 f16) from a torch-layout weight W[n][k] (row-major [N][K]):
// lane L -> col n0+(L&15); halves [0..7]=k kb..kb+7, [8..15]=kb+16..kb+23, kb = k0 + 8*(L>>4).
__device__ __forceinline__ v16h load_b_frag(const _Float16* W, int ldw, int n0,
                                            int k0, int lane) {
  int n = n0 + (lane & 15);
  const _Float16* p = W + (size_t)n * ldw + (k0 + 8 * (lane >> 4));
  v8h lo = *(const v8h*)(p);
  v8h hi = *(const v8h*)(p + 16);
  v16h r;
#pragma unroll
  for (int i = 0; i < 8; ++i) { r[i] = lo[i]; r[i + 8] = hi[i]; }
  return r;
}
__device__ __forceinline__ v8f wmma16(v16h a, v16h b, v8f c) {
  return __builtin_amdgcn_wmma_f32_16x16x32_f16(false, a, false, b, (short)0, c,
                                                false, false);
}

// ---------------- small utility kernels --------------------------------------
__global__ void k_f32_to_f16(const float* __restrict__ s, _Float16* __restrict__ d, int n) {
  int i = blockIdx.x * blockDim.x + threadIdx.x;
  if (i < n) d[i] = (_Float16)s[i];
}
__global__ void k_zero_f32(float* __restrict__ p, int n) {
  int i = blockIdx.x * blockDim.x + threadIdx.x;
  if (i < n) p[i] = 0.0f;
}

// ---------------- encoders layer 1 (tiny K) ----------------------------------
__global__ void k_edge_attr_enc1(const float* __restrict__ coords,
                                 const int* __restrict__ src, const int* __restrict__ dst,
                                 const float* __restrict__ W1, const float* __restrict__ b1,
                                 _Float16* __restrict__ mid, int E) {
  int idx = blockIdx.x * blockDim.x + threadIdx.x;
  int e = idx >> 7, j = idx & 127;
  if (e >= E) return;
  int s = src[e], d = dst[e];
  float rx = coords[3 * d + 0] - coords[3 * s + 0];
  float ry = coords[3 * d + 1] - coords[3 * s + 1];
  float rz = coords[3 * d + 2] - coords[3 * s + 2];
  float dist = sqrtf(rx * rx + ry * ry + rz * rz);
  const float* w = W1 + j * 4;
  float v = w[0] * rx + w[1] * ry + w[2] * rz + w[3] * dist + b1[j];
  mid[(size_t)e * HID + j] = (_Float16)gelu_exact(v);
}

__global__ void k_node_enc1(const float* __restrict__ x,
                            const float* __restrict__ W1, const float* __restrict__ b1,
                            _Float16* __restrict__ mid, int N) {
  int idx = blockIdx.x * blockDim.x + threadIdx.x;
  int nd = idx >> 7, j = idx & 127;
  if (nd >= N) return;
  const float* w = W1 + j * 3;
  const float* xx = x + nd * 3;
  float v = w[0] * xx[0] + w[1] * xx[1] + w[2] * xx[2] + b1[j];
  mid[(size_t)nd * HID + j] = (_Float16)gelu_exact(v);
}

// ---------------- generic GEMM: out[M x 128] = A[M x K] @ W[128 x K]^T + b ---
template <int K, bool GELU_ACT>
__global__ __launch_bounds__(256) void k_gemm128(const _Float16* __restrict__ A,
                                                 const _Float16* __restrict__ W,
                                                 const float* __restrict__ bias,
                                                 float* __restrict__ outF,
                                                 _Float16* __restrict__ outH, int M) {
  int tid = threadIdx.x, wave = tid >> 5, lane = tid & 31;
  int m0 = blockIdx.x * 64;
  v8f zero = {};
  v8f acc[4] = {zero, zero, zero, zero};
  for (int kt = 0; kt < K / 32; ++kt) {
    v16h b = load_b_frag(W, K, wave * 16, kt * 32, lane);
#pragma unroll
    for (int ms = 0; ms < 4; ++ms) {
      v16h a = load_a_frag(A, K, m0 + ms * 16, kt * 32, lane, M);
      acc[ms] = wmma16(a, b, acc[ms]);
    }
  }
  int n = wave * 16 + (lane & 15);
  float bb = bias[n];
#pragma unroll
  for (int ms = 0; ms < 4; ++ms)
#pragma unroll
    for (int r = 0; r < 8; ++r) {
      int m = m0 + ms * 16 + 8 * (lane >> 4) + r;
      if (m < M) {
        float v = acc[ms][r] + bb;
        if (GELU_ACT) v = gelu_exact(v);
        if (outF) outF[(size_t)m * HID + n] = v;
        if (outH) outH[(size_t)m * HID + n] = (_Float16)v;
      }
    }
}

// ---------------- fused MPNN message + scatter-aggregate ---------------------
// 64 edges per block; msg = MLP2([h_src|h_dst|e], W1[128x384], W2[128x128]);
// agg[dst] += msg ; cnt[dst] += 1.
union MsgSmem {
  _Float16 A[64][384];   // concat input tile
  _Float16 H[64][HID];   // GELU hidden tile (aliases A after all A reads)
};
__global__ __launch_bounds__(256) void k_mpnn_msg(
    const _Float16* __restrict__ h16, const _Float16* __restrict__ e16,
    const int* __restrict__ src, const int* __restrict__ dst,
    const _Float16* __restrict__ W1, const float* __restrict__ b1,
    const _Float16* __restrict__ W2, const float* __restrict__ b2,
    float* __restrict__ agg, float* __restrict__ cnt) {
  __shared__ MsgSmem sm;
  __shared__ int sDst[64];
  int tid = threadIdx.x;
  int e0 = blockIdx.x * 64;
  {  // stage concat tile via async gather->LDS: 4 threads/edge, 64B/segment
    int row = tid >> 2, part = tid & 3;
    int e = e0 + row;
    int s = src[e], d = dst[e];
    if (part == 0) sDst[row] = d;
    const v8h* ps = (const v8h*)(h16 + (size_t)s * HID + part * 32);
    const v8h* pd = (const v8h*)(h16 + (size_t)d * HID + part * 32);
    const v8h* pe = (const v8h*)(e16 + (size_t)e * HID + part * 32);
#pragma unroll
    for (int i = 0; i < 4; ++i) {
      async_ld_b128(&sm.A[row][part * 32 + i * 8],       ps + i);
      async_ld_b128(&sm.A[row][128 + part * 32 + i * 8], pd + i);
      async_ld_b128(&sm.A[row][256 + part * 32 + i * 8], pe + i);
    }
    __builtin_prefetch(e16 + (size_t)(e0 + 64) * HID, 0, 0);  // next tile
  }
  wait_async0();
  __syncthreads();
  int wave = tid >> 5, lane = tid & 31;
  v8f zero = {};
  v8f acc[4] = {zero, zero, zero, zero};
  for (int kt = 0; kt < 12; ++kt) {  // K = 384
    v16h b = load_b_frag(W1, 384, wave * 16, kt * 32, lane);
#pragma unroll
    for (int ms = 0; ms < 4; ++ms) {
      v16h a = load_a_frag(&sm.A[0][0], 384, ms * 16, kt * 32, lane, 1 << 30);
      acc[ms] = wmma16(a, b, acc[ms]);
    }
  }
  __syncthreads();  // all A reads done before aliased H writes
  int n = wave * 16 + (lane & 15);
  float bb = b1[n];
#pragma unroll
  for (int ms = 0; ms < 4; ++ms)
#pragma unroll
    for (int r = 0; r < 8; ++r) {
      int m = ms * 16 + 8 * (lane >> 4) + r;
      sm.H[m][n] = (_Float16)gelu_exact(acc[ms][r] + bb);
    }
  __syncthreads();
  v8f acc2[4] = {zero, zero, zero, zero};
  for (int kt = 0; kt < 4; ++kt) {  // K = 128
    v16h b = load_b_frag(W2, HID, wave * 16, kt * 32, lane);
#pragma unroll
    for (int ms = 0; ms < 4; ++ms) {
      v16h a = load_a_frag(&sm.H[0][0], HID, ms * 16, kt * 32, lane, 1 << 30);
      acc2[ms] = wmma16(a, b, acc2[ms]);
    }
  }
  float bb2 = b2[n];
#pragma unroll
  for (int ms = 0; ms < 4; ++ms)
#pragma unroll
    for (int r = 0; r < 8; ++r) {
      int m = ms * 16 + 8 * (lane >> 4) + r;
      atomicAdd(&agg[(size_t)sDst[m] * HID + n], acc2[ms][r] + bb2);
    }
  if (tid < 64) atomicAdd(&cnt[sDst[tid]], 1.0f);
}

// ---------------- fused MPNN node update: MLP2([h|agg/cnt]) + res + LN -------
union NodeSmem {
  _Float16 A[64][256];  // concat input tile
  float O[64][HID];     // pre-LN output (aliases A after all A reads)
};
__global__ __launch_bounds__(256) void k_mpnn_node(
    float* __restrict__ h, _Float16* __restrict__ h16,
    const float* __restrict__ agg, const float* __restrict__ cnt,
    const _Float16* __restrict__ W1, const float* __restrict__ b1,  // [128][256]
    const _Float16* __restrict__ W2, const float* __restrict__ b2,  // [128][128]
    const float* __restrict__ gg, const float* __restrict__ gb, int N) {
  __shared__ NodeSmem sm;
  __shared__ _Float16 Htile[64][HID];
  int tid = threadIdx.x;
  int m0 = blockIdx.x * 64;
  {
    int row = tid >> 2, part = tid & 3;
    int node = m0 + row; if (node >= N) node = N - 1;
    if (part < 2) {  // async gather of h16 row halves into LDS
      const v8h* p = (const v8h*)(h16 + (size_t)node * HID + part * 64);
#pragma unroll
      for (int i = 0; i < 8; ++i)
        async_ld_b128(&sm.A[row][part * 64 + i * 8], p + i);
    } else {
      int kb = (part - 2) * 64;
      float ic = 1.0f / (cnt[node] + 1e-8f);
      const float* pa = agg + (size_t)node * HID + kb;
      _Float16* q = &sm.A[row][HID + kb];
      for (int i = 0; i < 64; ++i) q[i] = (_Float16)(pa[i] * ic);
    }
  }
  wait_async0();
  __syncthreads();
  int wave = tid >> 5, lane = tid & 31;
  v8f zero = {};
  v8f acc[4] = {zero, zero, zero, zero};
  for (int kt = 0; kt < 8; ++kt) {  // K = 256
    v16h b = load_b_frag(W1, 256, wave * 16, kt * 32, lane);
#pragma unroll
    for (int ms = 0; ms < 4; ++ms) {
      v16h a = load_a_frag(&sm.A[0][0], 256, ms * 16, kt * 32, lane, 1 << 30);
      acc[ms] = wmma16(a, b, acc[ms]);
    }
  }
  __syncthreads();  // A dead -> O may alias
  int n = wave * 16 + (lane & 15);
  float bb = b1[n];
#pragma unroll
  for (int ms = 0; ms < 4; ++ms)
#pragma unroll
    for (int r = 0; r < 8; ++r) {
      int m = ms * 16 + 8 * (lane >> 4) + r;
      Htile[m][n] = (_Float16)gelu_exact(acc[ms][r] + bb);
    }
  __syncthreads();
  v8f acc2[4] = {zero, zero, zero, zero};
  for (int kt = 0; kt < 4; ++kt) {  // K = 128
    v16h b = load_b_frag(W2, HID, wave * 16, kt * 32, lane);
#pragma unroll
    for (int ms = 0; ms < 4; ++ms) {
      v16h a = load_a_frag(&Htile[0][0], HID, ms * 16, kt * 32, lane, 1 << 30);
      acc2[ms] = wmma16(a, b, acc2[ms]);
    }
  }
  float bb2 = b2[n];
#pragma unroll
  for (int ms = 0; ms < 4; ++ms)
#pragma unroll
    for (int r = 0; r < 8; ++r) {
      int m = ms * 16 + 8 * (lane >> 4) + r;
      int node = m0 + m; if (node >= N) node = N - 1;
      sm.O[m][n] = acc2[ms][r] + bb2 + h[(size_t)node * HID + n];
    }
  __syncthreads();
  if (tid < 64) {  // LayerNorm per row
    int node = m0 + tid;
    if (node < N) {
      float mean = 0.f;
      for (int k = 0; k < HID; ++k) mean += sm.O[tid][k];
      mean *= (1.0f / HID);
      float var = 0.f;
      for (int k = 0; k < HID; ++k) { float d = sm.O[tid][k] - mean; var += d * d; }
      var *= (1.0f / HID);
      float inv = rsqrtf(var + 1e-5f);
      for (int k = 0; k < HID; ++k) {
        float o = (sm.O[tid][k] - mean) * inv * gg[k] + gb[k];
        h[(size_t)node * HID + k] = o;
        h16[(size_t)node * HID + k] = (_Float16)o;
      }
    }
  }
}

// ---------------- global block ------------------------------------------------
__global__ void k_slice_softmax(const float* __restrict__ h,
                                const float* __restrict__ slW, const float* __restrict__ slB,
                                float* __restrict__ Wsm, int N) {
  int tid = threadIdx.x;
  int node = blockIdx.x * 8 + (tid >> 5);
  if (node >= N) return;
  int sl = tid & 31;
  const float* hr = h + (size_t)node * HID;
  const float* wr = slW + sl * HID;
  float s = slB[sl];
  for (int k = 0; k < HID; ++k) s += hr[k] * wr[k];
  float m = s;
  for (int o = 16; o > 0; o >>= 1) m = fmaxf(m, __shfl_xor(m, o, 32));
  float e = expf(s - m);
  float sum = e;
  for (int o = 16; o > 0; o >>= 1) sum += __shfl_xor(sum, o, 32);
  Wsm[(size_t)node * NSLICES + sl] = e / sum;
}

__global__ __launch_bounds__(256) void k_st_accum(const float* __restrict__ Wsm,
                                                  const float* __restrict__ h,
                                                  float* __restrict__ st, int N) {
  int tid = threadIdx.x;
  int s = (tid * 16) >> 7;   // constant slice per thread
  int kb = (tid * 16) & 127; // 16 contiguous features per thread
  float racc[16];
#pragma unroll
  for (int i = 0; i < 16; ++i) racc[i] = 0.f;
  int n0 = blockIdx.x * 128;
  int nend = n0 + 128; if (nend > N) nend = N;
  for (int node = n0; node < nend; ++node) {
    float w = Wsm[(size_t)node * NSLICES + s];
    const float* hr = h + (size_t)node * HID + kb;
#pragma unroll
    for (int i = 0; i < 16; ++i) racc[i] += w * hr[i];
  }
#pragma unroll
  for (int i = 0; i < 16; ++i) atomicAdd(&st[s * HID + kb + i], racc[i]);
}

__global__ __launch_bounds__(256) void k_glob_attn(
    const float* __restrict__ st, const float* __restrict__ inW, const float* __restrict__ inB,
    const float* __restrict__ outW, const float* __restrict__ outB,
    const float* __restrict__ ln1g, const float* __restrict__ ln1b,
    const float* __restrict__ fW1, const float* __restrict__ fb1,
    const float* __restrict__ fW2, const float* __restrict__ fb2,
    const float* __restrict__ ln2g, const float* __restrict__ ln2b,
    float* __restrict__ stOut) {
  __shared__ float sst[NSLICES][HID];
  __shared__ float big[NSLICES][4 * HID];  // qkv (cols 0..383) then ffn hidden
  __shared__ float ao[NSLICES][HID];
  __shared__ float tmp[NSLICES][HID];
  int tid = threadIdx.x;
  for (int i = tid; i < NSLICES * HID; i += 256) sst[i >> 7][i & 127] = st[i];
  __syncthreads();
  for (int i = tid; i < NSLICES * 3 * HID; i += 256) {  // qkv
    int s = i / (3 * HID), o = i % (3 * HID);
    float v = inB[o];
    const float* w = inW + (size_t)o * HID;
    for (int k = 0; k < HID; ++k) v += sst[s][k] * w[k];
    big[s][o] = v;
  }
  __syncthreads();
  {  // attention: thread = (slice, head); hd = 16
    int s = tid >> 3, hh = tid & 7;
    float sc[NSLICES];
    const float* q = &big[s][hh * 16];
    float mx = -1e30f;
    for (int t = 0; t < NSLICES; ++t) {
      const float* kk = &big[t][HID + hh * 16];
      float d = 0.f;
      for (int j = 0; j < 16; ++j) d += q[j] * kk[j];
      d *= 0.25f;  // 1/sqrt(16)
      sc[t] = d; mx = fmaxf(mx, d);
    }
    float sum = 0.f;
    for (int t = 0; t < NSLICES; ++t) { sc[t] = expf(sc[t] - mx); sum += sc[t]; }
    float inv = 1.f / sum;
    for (int j = 0; j < 16; ++j) {
      float a = 0.f;
      for (int t = 0; t < NSLICES; ++t) a += sc[t] * big[t][2 * HID + hh * 16 + j];
      ao[s][hh * 16 + j] = a * inv;
    }
  }
  __syncthreads();
  for (int i = tid; i < NSLICES * HID; i += 256) {  // out proj + residual
    int s = i >> 7, o = i & 127;
    float v = outB[o];
    const float* w = outW + (size_t)o * HID;
    for (int k = 0; k < HID; ++k) v += ao[s][k] * w[k];
    tmp[s][o] = sst[s][o] + v;
  }
  __syncthreads();
  if (tid < NSLICES) {  // LN1 -> sst
    float mean = 0.f;
    for (int k = 0; k < HID; ++k) mean += tmp[tid][k];
    mean *= (1.0f / HID);
    float var = 0.f;
    for (int k = 0; k < HID; ++k) { float d = tmp[tid][k] - mean; var += d * d; }
    float inv = rsqrtf(var * (1.0f / HID) + 1e-5f);
    for (int k = 0; k < HID; ++k)
      sst[tid][k] = (tmp[tid][k] - mean) * inv * ln1g[k] + ln1b[k];
  }
  __syncthreads();
  for (int i = tid; i < NSLICES * 4 * HID; i += 256) {  // ffn hidden (aliases qkv)
    int s = i / (4 * HID), o = i % (4 * HID);
    float v = fb1[o];
    const float* w = fW1 + (size_t)o * HID;
    for (int k = 0; k < HID; ++k) v += sst[s][k] * w[k];
    big[s][o] = gelu_exact(v);
  }
  __syncthreads();
  for (int i = tid; i < NSLICES * HID; i += 256) {  // ffn out + residual
    int s = i >> 7, o = i & 127;
    float v = fb2[o];
    const float* w = fW2 + (size_t)o * 4 * HID;
    for (int k = 0; k < 4 * HID; ++k) v += big[s][k] * w[k];
    tmp[s][o] = sst[s][o] + v;
  }
  __syncthreads();
  if (tid < NSLICES) {  // LN2 -> stOut
    float mean = 0.f;
    for (int k = 0; k < HID; ++k) mean += tmp[tid][k];
    mean *= (1.0f / HID);
    float var = 0.f;
    for (int k = 0; k < HID; ++k) { float d = tmp[tid][k] - mean; var += d * d; }
    float inv = rsqrtf(var * (1.0f / HID) + 1e-5f);
    for (int k = 0; k < HID; ++k)
      stOut[tid * HID + k] = (tmp[tid][k] - mean) * inv * ln2g[k] + ln2b[k];
  }
}

__global__ __launch_bounds__(256) void k_wsm_apply(float* __restrict__ h,
                                                   _Float16* __restrict__ h16,
                                                   const float* __restrict__ Wsm,
                                                   const float* __restrict__ stF, int N) {
  __shared__ float sst[NSLICES * HID];
  int tid = threadIdx.x;
  for (int i = tid; i < NSLICES * HID; i += 256) sst[i] = stF[i];
  __syncthreads();
  size_t g = (size_t)blockIdx.x * 256 + tid;
  if (g >= (size_t)N * HID) return;
  int node = (int)(g >> 7), k = (int)(g & 127);
  float v = h[g];
  const float* wr = Wsm + (size_t)node * NSLICES;
#pragma unroll
  for (int s = 0; s < NSLICES; ++s) v += wr[s] * sst[s * HID + k];
  h[g] = v;
  h16[g] = (_Float16)v;
}

// ---------------- decoder -----------------------------------------------------
__global__ __launch_bounds__(128) void k_decoder(
    const float* __restrict__ h, const float* __restrict__ lng, const float* __restrict__ lnb,
    const float* __restrict__ W1, const float* __restrict__ b1,   // [64][128]
    const float* __restrict__ W2, const float* __restrict__ b2,   // [9][64]
    float* __restrict__ out, int N) {
  __shared__ float t[HID];
  __shared__ float red[HID];
  __shared__ float hid[64];
  int node = blockIdx.x;
  int tid = threadIdx.x;
  float x = h[(size_t)node * HID + tid];
  red[tid] = x; __syncthreads();
  for (int o = 64; o > 0; o >>= 1) { if (tid < o) red[tid] += red[tid + o]; __syncthreads(); }
  float mean = red[0] * (1.0f / HID); __syncthreads();
  float dx = x - mean;
  red[tid] = dx * dx; __syncthreads();
  for (int o = 64; o > 0; o >>= 1) { if (tid < o) red[tid] += red[tid + o]; __syncthreads(); }
  float inv = rsqrtf(red[0] * (1.0f / HID) + 1e-5f);
  t[tid] = dx * inv * lng[tid] + lnb[tid];
  __syncthreads();
  if (tid < 64) {
    const float* w = W1 + tid * HID;
    float v = b1[tid];
    for (int k = 0; k < HID; ++k) v += w[k] * t[k];
    hid[tid] = gelu_exact(v);
  }
  __syncthreads();
  if (tid < 9) {
    const float* w = W2 + tid * 64;
    float v = b2[tid];
    for (int k = 0; k < 64; ++k) v += w[k] * hid[k];
    out[(size_t)node * 9 + tid] = v;
  }
}

// =============================================================================
extern "C" void kernel_launch(void* const* d_in, const int* in_sizes, int n_in,
                              void* d_out, int out_size, void* d_ws, size_t ws_size,
                              hipStream_t stream) {
  (void)in_sizes; (void)n_in; (void)out_size; (void)ws_size;
  const int N = 10000, E = 160000;
  const float* x      = (const float*)d_in[0];
  const float* coords = (const float*)d_in[1];
  const int*   ei     = (const int*)d_in[2];
  const int* src = ei;
  const int* dst = ei + E;

  // params leaves at d_in[3..], jax tree_flatten order (dict keys sorted, lists in order)
  void* const* P = d_in + 3;
  enum {
    DEC_W1 = 0, DEC_W2, DEC_B1, DEC_B2, DEC_LNB, DEC_LNG,                 // dec
    EE_W1, EE_B1, EE_W2, EE_B2,                                           // edge_enc
    G_FW1, G_FB1, G_FW2, G_FB2, G_INW, G_INB, G_LN1B, G_LN1G,
    G_LN2B, G_LN2G, G_OUTW, G_OUTB, G_SLW, G_SLB,                         // glob (sorted)
    NE_W1, NE_B1, NE_W2, NE_B2,                                           // node_enc
    POST0                                                                 // post[0] starts here
  };
  const int PRE0 = POST0 + 30;  // 3 post layers x 10 leaves each, then pre
  // per-mpnn-layer leaf offsets: em_W1,em_b1,em_W2,em_b2,nm_W1,nm_b1,nm_W2,nm_b2,norm_b,norm_g
  auto F = [&](int i) { return (const float*)P[i]; };

  // ---- workspace carve-up ----
  char* ws = (char*)d_ws;
  size_t off = 0;
  auto alloc = [&](size_t bytes) -> char* {
    char* p = ws + off;
    off = (off + bytes + 255) & ~(size_t)255;
    return p;
  };
  float*     h    = (float*)alloc((size_t)N * HID * 4);
  _Float16*  h16  = (_Float16*)alloc((size_t)N * HID * 2);
  _Float16*  e16  = (_Float16*)alloc((size_t)E * HID * 2);
  _Float16*  mid  = (_Float16*)alloc((size_t)E * HID * 2);  // enc hidden (edge then node)
  float*     agg  = (float*)alloc((size_t)N * HID * 4);
  float*     cnt  = (float*)alloc((size_t)N * 4);
  float*     Wsm  = (float*)alloc((size_t)N * NSLICES * 4);
  float*     st   = (float*)alloc(NSLICES * HID * 4);
  float*     stF  = (float*)alloc(NSLICES * HID * 4);
  _Float16*  eeW2h = (_Float16*)alloc(128 * 128 * 2);
  _Float16*  neW2h = (_Float16*)alloc(128 * 128 * 2);
  _Float16*  emW1h[6]; _Float16* emW2h[6]; _Float16* nmW1h[6]; _Float16* nmW2h[6];
  for (int l = 0; l < 6; ++l) {
    emW1h[l] = (_Float16*)alloc(128 * 384 * 2);
    emW2h[l] = (_Float16*)alloc(128 * 128 * 2);
    nmW1h[l] = (_Float16*)alloc(128 * 256 * 2);
    nmW2h[l] = (_Float16*)alloc(128 * 128 * 2);
  }

  auto cvt = [&](const float* s, _Float16* d, int n) {
    k_f32_to_f16<<<(n + 255) / 256, 256, 0, stream>>>(s, d, n);
  };
  // ---- weight conversion to f16 (torch [out][in] layout preserved) ----
  cvt(F(EE_W2), eeW2h, 128 * 128);
  cvt(F(NE_W2), neW2h, 128 * 128);
  int layerBase[6];
  for (int l = 0; l < 6; ++l) {
    layerBase[l] = (l < 3) ? (PRE0 + l * 10) : (POST0 + (l - 3) * 10);
    cvt(F(layerBase[l] + 0), emW1h[l], 128 * 384);
    cvt(F(layerBase[l] + 2), emW2h[l], 128 * 128);
    cvt(F(layerBase[l] + 4), nmW1h[l], 128 * 256);
    cvt(F(layerBase[l] + 6), nmW2h[l], 128 * 128);
  }

  // ---- encoders ----
  k_edge_attr_enc1<<<(E * HID) / 256, 256, 0, stream>>>(coords, src, dst, F(EE_W1), F(EE_B1), mid, E);
  k_gemm128<128, false><<<E / 64, 256, 0, stream>>>(mid, eeW2h, F(EE_B2), nullptr, e16, E);
  k_node_enc1<<<((N * HID) + 255) / 256, 256, 0, stream>>>(x, F(NE_W1), F(NE_B1), mid, N);
  k_gemm128<128, false><<<(N + 63) / 64, 256, 0, stream>>>(mid, neW2h, F(NE_B2), h, h16, N);

  // ---- MPNN layer runner ----
  auto mpnn = [&](int l) {
    int b = layerBase[l];
    k_zero_f32<<<((N * HID) + 255) / 256, 256, 0, stream>>>(agg, N * HID);
    k_zero_f32<<<(N + 255) / 256, 256, 0, stream>>>(cnt, N);
    k_mpnn_msg<<<E / 64, 256, 0, stream>>>(h16, e16, src, dst,
                                           emW1h[l], F(b + 1), emW2h[l], F(b + 3),
                                           agg, cnt);
    k_mpnn_node<<<(N + 63) / 64, 256, 0, stream>>>(h, h16, agg, cnt,
                                                   nmW1h[l], F(b + 5), nmW2h[l], F(b + 7),
                                                   F(b + 9), F(b + 8), N);  // norm_g, norm_b
  };

  for (int l = 0; l < 3; ++l) mpnn(l);  // pre layers

  // ---- global physics-attention block ----
  k_slice_softmax<<<(N + 7) / 8, 256, 0, stream>>>(h, F(G_SLW), F(G_SLB), Wsm, N);
  k_zero_f32<<<(NSLICES * HID + 255) / 256, 256, 0, stream>>>(st, NSLICES * HID);
  k_st_accum<<<(N + 127) / 128, 256, 0, stream>>>(Wsm, h, st, N);
  k_glob_attn<<<1, 256, 0, stream>>>(st, F(G_INW), F(G_INB), F(G_OUTW), F(G_OUTB),
                                     F(G_LN1G), F(G_LN1B), F(G_FW1), F(G_FB1),
                                     F(G_FW2), F(G_FB2), F(G_LN2G), F(G_LN2B), stF);
  k_wsm_apply<<<((N * HID) + 255) / 256, 256, 0, stream>>>(h, h16, Wsm, stF, N);

  for (int l = 3; l < 6; ++l) mpnn(l);  // post layers

  // ---- decoder ----
  k_decoder<<<N, 128, 0, stream>>>(h, F(DEC_LNG), F(DEC_LNB), F(DEC_W1), F(DEC_B1),
                                   F(DEC_W2), F(DEC_B2), (float*)d_out, N);
}
